// FeedBack_10625749090780
// MI455X (gfx1250) — matrix operand order repaired and checked by reading, not compiled, not run on hardware
//
#include <hip/hip_runtime.h>
#include <hip/hip_bf16.h>

#define UNITS 1024
#define OUT_STEPS 32
#define NUM_FEATURES 4
#define BATCH 1024
#define IN_STEPS 64
#define GCOLS (4 * UNITS)          // 4096 gate columns (i|f|g|o)
#define KTILES (UNITS / 32)        // 32 K-tiles of 32
#define CTILES (GCOLS / 16)        // 256 column tiles of 16

typedef __bf16 bf16_t;
typedef __attribute__((ext_vector_type(16))) __bf16 v16bf;
typedef __attribute__((ext_vector_type(8)))  __bf16 v8bf;
typedef __attribute__((ext_vector_type(8)))  float  v8f;

__device__ __forceinline__ float fast_tanh(float x) {
#if __has_builtin(__builtin_amdgcn_tanhf)
    return __builtin_amdgcn_tanhf(x);            // native v_tanh_f32
#elif __has_builtin(__builtin_amdgcn_tanh_f32)
    return __builtin_amdgcn_tanh_f32(x);
#else
    return tanhf(x);
#endif
}
__device__ __forceinline__ float sigmoid_f(float x) {
    // exact identity: sigma(x) = 0.5*tanh(x/2) + 0.5  (single trans op)
    return __builtin_fmaf(0.5f, fast_tanh(0.5f * x), 0.5f);
}

// ---------------------------------------------------------------------------
// Pack U (fp32, row-major [K=1024][N=4096]) into bf16 WMMA B-fragment order.
// Tile = 32(K) x 16(N). Lane L (0..15) owns column N=L, K = kBase+0..15;
// lane L+16 owns column N=L, K = kBase+16..31. Each lane's 16 elements are
// contiguous (32B) -> two b128 loads per fragment.
// ---------------------------------------------------------------------------
__global__ __launch_bounds__(256) void pack_U_kernel(
        const float* __restrict__ U, bf16_t* __restrict__ Upk) {
    int idx = blockIdx.x * 256 + threadIdx.x;       // 0 .. 4M-1
    int j       = idx & 15;
    int lane    = (idx >> 4) & 31;
    int colTile = (idx >> 9) & (CTILES - 1);
    int kt      = idx >> 17;
    int n = colTile * 16 + (lane & 15);
    int k = kt * 32 + ((lane >> 4) * 16) + j;
    Upk[idx] = (bf16_t)U[k * GCOLS + n];
}

// ---------------------------------------------------------------------------
// Async-stage one K-tile's worth of B (8 tiles x 1KB = 8KB) into LDS.
// 256 threads x 2 async b128 each. LDS tile tb = g*2 + ut at offset tb*1024.
// ---------------------------------------------------------------------------
__device__ __forceinline__ void loadB_async(
        const bf16_t* __restrict__ Upk, int kt, int unitBlock,
        uint32_t ldsBufAddr /* LDS byte address of buffer base */) {
#pragma unroll
    for (int i = 0; i < 2; ++i) {
        const int c  = threadIdx.x * 2 + i;     // chunk 0..511 (16B each)
        const int tb = c >> 6;                  // 0..7
        const int g  = tb >> 1;
        const int ut = tb & 1;
        const int colTile = g * (UNITS / 16) + unitBlock * 2 + ut;
        const char* src = (const char*)Upk
                        + ((size_t)(kt * CTILES + colTile) << 10)
                        + ((size_t)(c & 63) << 4);
        const uint32_t dst = ldsBufAddr + (uint32_t)(c << 4);
        asm volatile("global_load_async_to_lds_b128 %0, %1, off"
                     :: "v"(dst), "v"((unsigned long long)(uintptr_t)src)
                     : "memory");
    }
}

// ---------------------------------------------------------------------------
// Fused LSTM step:  Z = h@U (+ x@W + b in epilogue), gates, c/h update.
// Grid: (8 row-blocks, 32 unit-blocks). Block = 256 threads = 8 waves,
// arranged as 4 row-groups x 2 unit-groups. Wave tile: 32 rows x 16 units
// x all 4 gates (2 A-frags x 4 B-frags -> 8 WMMAs per K-tile).
// B staged in LDS via async-to-LDS, triple buffered, 1 barrier / K-tile.
// ---------------------------------------------------------------------------
__global__ __launch_bounds__(256) void lstm_step_kernel(
        const bf16_t* __restrict__ hIn,    // [1024][1024] bf16 (prev h)
        const bf16_t* __restrict__ Upk,    // packed U bf16
        const float*  __restrict__ W,      // [4][4096]
        const float*  __restrict__ bias,   // [4096]
        const float*  __restrict__ xt,     // x input: xt[row*xStride + f]
        int xStride,
        float*        __restrict__ cState, // [1024][1024] fp32 (in place)
        bf16_t*       __restrict__ hOutB,  // [1024][1024] bf16 (next h)
        float*        __restrict__ hOutF)  // [1024][1024] fp32 (for h@Wd)
{
    __shared__ __align__(16) bf16_t shB[3 * 8 * 512];   // 3 x 8KB

    const int wave = threadIdx.x >> 5;                  // 0..7
    const int lane = threadIdx.x & 31;
    const int rowGroup = wave & 3;                      // 4 row groups
    const int colGroup = wave >> 2;                     // 2 unit groups
    const int rowBase   = blockIdx.x * 128 + rowGroup * 32;
    const int unitBlock = blockIdx.y;
    const int unitBase  = unitBlock * 32;

    const uint32_t shBase = (uint32_t)(uintptr_t)&shB[0];

    v8f acc[2][4];                                      // [rowTile][gate]
#pragma unroll
    for (int rt = 0; rt < 2; ++rt)
#pragma unroll
        for (int g = 0; g < 4; ++g)
            acc[rt][g] = (v8f){0.f,0.f,0.f,0.f,0.f,0.f,0.f,0.f};

    const int kSel = (lane < 16) ? 0 : 16;              // A-frag half select

    // Prologue: stage K-tile 0 into buffer 0.
    loadB_async(Upk, 0, unitBlock, shBase);

    for (int kt = 0; kt < KTILES; ++kt) {
        const int buf = kt % 3;
        if (kt + 1 < KTILES) {
            loadB_async(Upk, kt + 1, unitBlock, shBase + ((kt + 1) % 3) * 8192u);
            asm volatile("s_wait_asynccnt 2" ::: "memory");   // batch kt done
        } else {
            asm volatile("s_wait_asynccnt 0" ::: "memory");
        }
        // Single barrier per K-tile: triple buffering guarantees the batch
        // issued above targets a buffer no wave can still be reading.
        __syncthreads();

        const int kBase = kt * 32;
        // A fragments (2 row tiles), direct from row-major bf16 h.
        v16bf afrag[2];
#pragma unroll
        for (int rt = 0; rt < 2; ++rt) {
            const int laneRow = rowBase + rt * 16 + (lane & 15);
            const bf16_t* ap = hIn + laneRow * UNITS + kBase + kSel;
            v8bf alo = *(const v8bf*)(ap);
            v8bf ahi = *(const v8bf*)(ap + 8);
            afrag[rt] = __builtin_shufflevector(alo, ahi,
                0,1,2,3,4,5,6,7,8,9,10,11,12,13,14,15);
        }
#pragma unroll
        for (int g = 0; g < 4; ++g) {
            const int tb = g * 2 + colGroup;
            const bf16_t* bp = &shB[buf * 4096 + tb * 512 + lane * 16];
            v8bf blo = *(const v8bf*)(bp);
            v8bf bhi = *(const v8bf*)(bp + 8);
            v16bf bfrag = __builtin_shufflevector(blo, bhi,
                0,1,2,3,4,5,6,7,8,9,10,11,12,13,14,15);
#pragma unroll
            for (int rt = 0; rt < 2; ++rt) {
                acc[rt][g] = __builtin_amdgcn_wmma_f32_16x16x32_bf16(
                    false, afrag[rt], false, bfrag,
                    (short)0, acc[rt][g], false, false);
            }
        }
    }

    // Epilogue: + x@W + b, gates, state update. C/D layout (ISA 7.12.2):
    // v8f element e -> row = base + e + 8*(lane>>4), col = (lane&15).
    const int hi   = lane >> 4;
    const int nCol = lane & 15;
    const int ucol = unitBase + colGroup * 16 + nCol;

    float wv[4][4], bv[4];
#pragma unroll
    for (int g = 0; g < 4; ++g) {
        const int zc = g * UNITS + ucol;
        bv[g] = bias[zc];
#pragma unroll
        for (int f = 0; f < NUM_FEATURES; ++f)
            wv[g][f] = W[f * GCOLS + zc];
    }
#pragma unroll
    for (int rt = 0; rt < 2; ++rt) {
#pragma unroll
        for (int e = 0; e < 8; ++e) {
            const int row = rowBase + rt * 16 + hi * 8 + e;
            const float4 xv = *(const float4*)(xt + row * xStride);
            float z[4];
#pragma unroll
            for (int g = 0; g < 4; ++g)
                z[g] = acc[rt][g][e] + bv[g]
                     + xv.x * wv[g][0] + xv.y * wv[g][1]
                     + xv.z * wv[g][2] + xv.w * wv[g][3];
            const float ig = sigmoid_f(z[0]);
            const float fg = sigmoid_f(z[1]);
            const float gg = fast_tanh(z[2]);
            const float og = sigmoid_f(z[3]);
            const int off  = row * UNITS + ucol;
            const float cN = fg * cState[off] + ig * gg;
            const float hN = og * fast_tanh(cN);
            cState[off] = cN;
            hOutF[off]  = hN;
            hOutB[off]  = (bf16_t)hN;
        }
    }
}

// ---------------------------------------------------------------------------
// Projection: pred = h @ Wd + bd  (N=4). One thread per batch row,
// K vectorized by 4.
// ---------------------------------------------------------------------------
__global__ __launch_bounds__(256) void pred_kernel(
        const float* __restrict__ hF,   // [1024][1024]
        const float* __restrict__ Wd,   // [1024][4]
        const float* __restrict__ bd,   // [4]
        float* __restrict__ outBase,    // d_out + step*4, row stride OUT_STEPS*4
        float* __restrict__ predbuf)    // [1024][4]
{
    const int row = blockIdx.x * 256 + threadIdx.x;   // 0..1023
    float4 s = *(const float4*)bd;
    const float* hrow = hF + row * UNITS;
    for (int k = 0; k < UNITS; k += 4) {
        const float4 h = *(const float4*)(hrow + k);
        const float4 w0 = *(const float4*)(Wd + (k + 0) * 4);
        const float4 w1 = *(const float4*)(Wd + (k + 1) * 4);
        const float4 w2 = *(const float4*)(Wd + (k + 2) * 4);
        const float4 w3 = *(const float4*)(Wd + (k + 3) * 4);
        s.x += h.x * w0.x + h.y * w1.x + h.z * w2.x + h.w * w3.x;
        s.y += h.x * w0.y + h.y * w1.y + h.z * w2.y + h.w * w3.y;
        s.z += h.x * w0.z + h.y * w1.z + h.z * w2.z + h.w * w3.z;
        s.w += h.x * w0.w + h.y * w1.w + h.z * w2.w + h.w * w3.w;
    }
    *(float4*)(outBase + row * (OUT_STEPS * 4)) = s;
    *(float4*)(predbuf + row * 4) = s;
}

// ---------------------------------------------------------------------------
extern "C" void kernel_launch(void* const* d_in, const int* in_sizes, int n_in,
                              void* d_out, int out_size, void* d_ws, size_t ws_size,
                              hipStream_t stream) {
    const float* x  = (const float*)d_in[0];   // [1024][64][4]
    const float* W  = (const float*)d_in[1];   // [4][4096]
    const float* U  = (const float*)d_in[2];   // [1024][4096]
    const float* b  = (const float*)d_in[3];   // [4096]
    const float* Wd = (const float*)d_in[4];   // [1024][4]
    const float* bd = (const float*)d_in[5];   // [4]
    float* out = (float*)d_out;                // [1024][32][4]

    char* ws = (char*)d_ws;
    bf16_t* Upk = (bf16_t*)(ws);                          // 8 MB
    bf16_t* hbA = (bf16_t*)(ws + (8u  << 20));            // 2 MB
    bf16_t* hbB = (bf16_t*)(ws + (10u << 20));            // 2 MB
    float*  cSt = (float*) (ws + (12u << 20));            // 4 MB
    float*  hF  = (float*) (ws + (16u << 20));            // 4 MB
    float*  pb  = (float*) (ws + (20u << 20));            // 16 KB

    hipMemsetAsync(hbA, 0, (size_t)BATCH * UNITS * sizeof(bf16_t), stream);
    hipMemsetAsync(cSt, 0, (size_t)BATCH * UNITS * sizeof(float), stream);

    pack_U_kernel<<<(UNITS * GCOLS) / 256, 256, 0, stream>>>(U, Upk);

    const dim3 grid(8, 32);
    // 95 total steps: tau 0..63 warmup (input = x[:,tau,:]),
    // tau 64..94 decode (input = previous prediction).
    for (int tau = 0; tau < IN_STEPS + OUT_STEPS - 1; ++tau) {
        const bf16_t* hin  = (tau & 1) ? hbB : hbA;
        bf16_t*       hout = (tau & 1) ? hbA : hbB;
        const float* xtp;
        int xs;
        if (tau < IN_STEPS) { xtp = x + tau * NUM_FEATURES; xs = IN_STEPS * NUM_FEATURES; }
        else                { xtp = pb;                     xs = NUM_FEATURES; }
        lstm_step_kernel<<<grid, 256, 0, stream>>>(
            hin, Upk, W, b, xtp, xs, cSt, hout, hF);
        if (tau >= IN_STEPS - 1) {
            const int s = tau - (IN_STEPS - 1);          // 0..31
            pred_kernel<<<BATCH / 256, 256, 0, stream>>>(
                hF, Wd, bd, out + s * NUM_FEATURES, pb);
        }
    }
}
